// AttnNet_12695923327081
// MI455X (gfx1250) — matrix operand order repaired and checked by reading, not compiled, not run on hardware
//
#include <hip/hip_runtime.h>
#include <hip/hip_bf16.h>

typedef __attribute__((ext_vector_type(16))) _Float16 v16h;
typedef __attribute__((ext_vector_type(8)))  float    v8f;
typedef __attribute__((ext_vector_type(4)))  float    v4f;

// ---------------------------------------------------------------------------
// Phase 1: partial energies via v_wmma_f32_16x16x32_f16.
// Grid: one wave32 workgroup per (b, s_tile16, h_chunk). NC=4 h-chunks of 1024.
// part layout: [b][s_tile(32)][hc(4)][m(16)]  (float)
//
// B fragment replicates q across all 16 N columns (independent VALU movs that
// also cover the WMMA->VALU hazard window), so every column of C holds the
// energies. Two accumulators break the WMMA->WMMA RAW chain on C.
// ---------------------------------------------------------------------------
__global__ __launch_bounds__(32)
void attn_energies_wmma(const float* __restrict__ q,      // [B,H]
                        const float* __restrict__ facts,  // [B,S,H]
                        float* __restrict__ part)         // [B,32,4,16]
{
    constexpr int S = 512, H = 4096;
    constexpr int NC = 4, HC = H / NC;        // 1024 h per wave

    const int wg   = blockIdx.x;              // 64*32*4 workgroups
    const int hc   = wg & (NC - 1);
    const int st   = (wg >> 2) & 31;          // s-tile index
    const int b    = wg >> 7;
    const int lane = threadIdx.x;             // 0..31 (wave32)
    const int m    = lane & 15;               // A-fragment row for this lane
    const int hi   = (lane < 16) ? 0 : 8;     // first K-run offset per ISA A-layout

    const float* qrow = q + (size_t)b * H + hc * HC;
    const float* arow = facts + (((size_t)b * S + st * 16 + m) * H) + hc * HC + hi;

    v8f acc0 = {};
    v8f acc1 = {};

    for (int k = 0; k < HC; k += 64) {
        #pragma unroll
        for (int u = 0; u < 2; ++u) {
            const int kk = k + u * 32;

            // Prefetch the stream ~1KB ahead (global_prefetch_b8).
            __builtin_prefetch(arow + kk + 256, 0, 0);

            // B fragment: element i = q[kk + hi + i] (i<8), q[kk + hi + i + 8] (i>=8), for all 16 n.
            v16h bfrag;
            #pragma unroll
            for (int i = 0; i < 16; ++i) bfrag[i] = (_Float16)qrow[kk + hi + (i < 8 ? i : i + 8)];

            // A fragment: halves 0..7 = K {hi..hi+7}, halves 8..15 = K {hi+16..hi+23}.
            const v4f* p4 = (const v4f*)(arow + kk);
            const v4f f0 = p4[0];   // K hi+0..3
            const v4f f1 = p4[1];   // K hi+4..7
            const v4f f2 = p4[4];   // K hi+16..19
            const v4f f3 = p4[5];   // K hi+20..23
            v16h afrag;
            #pragma unroll
            for (int i = 0; i < 4; ++i) {
                afrag[i]      = (_Float16)f0[i];
                afrag[4 + i]  = (_Float16)f1[i];
                afrag[8 + i]  = (_Float16)f2[i];
                afrag[12 + i] = (_Float16)f3[i];
            }

            if (u == 0) {
                acc0 = __builtin_amdgcn_wmma_f32_16x16x32_f16(
                    false, afrag, false, bfrag, (short)0, acc0, false, false);
            } else {
                acc1 = __builtin_amdgcn_wmma_f32_16x16x32_f16(
                    false, afrag, false, bfrag, (short)0, acc1, false, false);
            }
        }
    }

    const v8f acc = acc0 + acc1;

    // C layout: VGPR r holds M=r (lanes 0-15) / M=r+8 (lanes 16-31); all N equal.
    float* dst = part + ((((size_t)b * 32 + st) * NC) + hc) * 16;
    if (lane == 0) {
        #pragma unroll
        for (int r = 0; r < 8; ++r) dst[r] = acc[r];
    } else if (lane == 16) {
        #pragma unroll
        for (int r = 0; r < 8; ++r) dst[8 + r] = acc[r];
    }
}

// ---------------------------------------------------------------------------
// Phase 2: reduce 4 partials per (b,s), softmax over S=512 per batch.
// Deterministic fixed-order LDS tree reductions.
// ---------------------------------------------------------------------------
__global__ __launch_bounds__(512)
void attn_softmax(const float* __restrict__ part,  // [B,32,4,16]
                  float* __restrict__ out)         // [B,1,S]
{
    constexpr int S = 512, NC = 4;
    const int b = blockIdx.x;
    const int s = threadIdx.x;            // 0..511
    const int st = s >> 4, m = s & 15;

    const float* p = part + (((size_t)b * 32 + st) * NC) * 16 + m;
    const float e = p[0] + p[16] + p[32] + p[48];

    __shared__ float sm[S];
    sm[s] = e;
    __syncthreads();
    for (int off = S / 2; off > 0; off >>= 1) {
        if (s < off) sm[s] = fmaxf(sm[s], sm[s + off]);
        __syncthreads();
    }
    const float mx = sm[0];
    __syncthreads();

    const float ex = __expf(e - mx);
    sm[s] = ex;
    __syncthreads();
    for (int off = S / 2; off > 0; off >>= 1) {
        if (s < off) sm[s] += sm[s + off];
        __syncthreads();
    }
    out[(size_t)b * S + s] = ex * (1.0f / sm[0]);
}

// ---------------------------------------------------------------------------
extern "C" void kernel_launch(void* const* d_in, const int* in_sizes, int n_in,
                              void* d_out, int out_size, void* d_ws, size_t ws_size,
                              hipStream_t stream) {
    const float* questions = (const float*)d_in[0];  // [64, 4096]
    const float* facts     = (const float*)d_in[1];  // [64, 512, 4096]
    float*       out       = (float*)d_out;          // [64, 1, 512]
    float*       part      = (float*)d_ws;           // 64*32*4*16 floats = 512 KB

    dim3 grid1(64 * 32 * 4);
    attn_energies_wmma<<<grid1, 32, 0, stream>>>(questions, facts, part);
    attn_softmax<<<64, 512, 0, stream>>>(part, out);
}
